// Encoder_22239340658989
// MI455X (gfx1250) — compile-verified, matrix operand order checked by reading, not run on hardware
//
#include <hip/hip_runtime.h>

typedef _Float16 v4h  __attribute__((ext_vector_type(4)));
typedef _Float16 v8h  __attribute__((ext_vector_type(8)));
typedef _Float16 v16h __attribute__((ext_vector_type(16)));
typedef float    v8f  __attribute__((ext_vector_type(8)));
typedef int      v4i  __attribute__((ext_vector_type(4)));

#define GLOBAL_AS __attribute__((address_space(1)))
#define LDS_AS    __attribute__((address_space(3)))

#if defined(__has_builtin)
#if __has_builtin(__builtin_amdgcn_global_load_async_to_lds_b128)
#define HAS_ASYNC 1
#else
#define HAS_ASYNC 0
#endif
#if __has_builtin(__builtin_amdgcn_s_wait_asynccnt)
#define WAIT_ASYNC() __builtin_amdgcn_s_wait_asynccnt(0)
#else
#define WAIT_ASYNC() asm volatile("s_wait_asynccnt 0x0" ::: "memory")
#endif
#if __has_builtin(__builtin_amdgcn_update_dpp)
#define HAS_DPP 1
#else
#define HAS_DPP 0
#endif
#else
#define HAS_ASYNC 0
#define HAS_DPP 0
#define WAIT_ASYNC()
#endif

static __device__ __forceinline__ v16h cat16(v8h lo, v8h hi) {
    return __builtin_shufflevector(lo, hi, 0,1,2,3,4,5,6,7,8,9,10,11,12,13,14,15);
}

// 16-byte global -> LDS copy: async (CDNA5 GLOBAL_LOAD_ASYNC_TO_LDS_B128) or fallback.
static __device__ __forceinline__ void cp16(const _Float16* g, _Float16* l) {
#if HAS_ASYNC
    __builtin_amdgcn_global_load_async_to_lds_b128(
        (GLOBAL_AS v4i*)(g), (LDS_AS v4i*)(l), 0, 0);
#else
    *reinterpret_cast<v8h*>(l) = *reinterpret_cast<const v8h*>(g);
#endif
}

static __device__ __forceinline__ void stage_fence() {
#if HAS_ASYNC
    WAIT_ASYNC();
#endif
    __syncthreads();
}

// ---- 16-lane row reductions (match WMMA C-layout halves) via DPP row_ror ----
#if HAS_DPP
template <int CTRL>
static __device__ __forceinline__ float dpp_rot(float x) {
    const int i = __float_as_int(x);
    return __int_as_float(__builtin_amdgcn_update_dpp(i, i, CTRL, 0xF, 0xF, false));
}
#endif
static __device__ __forceinline__ float row16_max(float v) {
#if HAS_DPP
    v = fmaxf(v, dpp_rot<0x121>(v));   // row_ror:1
    v = fmaxf(v, dpp_rot<0x122>(v));   // row_ror:2
    v = fmaxf(v, dpp_rot<0x124>(v));   // row_ror:4
    v = fmaxf(v, dpp_rot<0x128>(v));   // row_ror:8
#else
    #pragma unroll
    for (int m = 1; m < 16; m <<= 1) v = fmaxf(v, __shfl_xor(v, m, 32));
#endif
    return v;
}
static __device__ __forceinline__ float row16_sum(float v) {
#if HAS_DPP
    v += dpp_rot<0x121>(v);
    v += dpp_rot<0x122>(v);
    v += dpp_rot<0x124>(v);
    v += dpp_rot<0x128>(v);
#else
    #pragma unroll
    for (int m = 1; m < 16; m <<= 1) v += __shfl_xor(v, m, 32);
#endif
    return v;
}

// ---------------------------------------------------------------------------
// WMMA GEMM, all-f16 operands: C[M,N] = act(A[M,K] * BT[N,K]^T + bias)
// A row-major [M][K] f16, BT row-major [N][K] f16 (pre-transposed weights).
// 128x128 block tile, BK=64, double-buffered async LDS staging.
// ---------------------------------------------------------------------------
template <bool RELU, typename OT>
__global__ __launch_bounds__(256)
void wmma_gemm_f16(const _Float16* __restrict__ A, const _Float16* __restrict__ BT,
                   const float* __restrict__ bias, OT* __restrict__ C,
                   int M, int N, int K)
{
    constexpr int BM = 128, BN = 128, BK = 64, PADH = 72; // 144B rows (16B multiple)
    __shared__ alignas(16) _Float16 As[2][BM][PADH];
    __shared__ alignas(16) _Float16 Bs[2][BN][PADH];

    const int tid  = threadIdx.x;
    const int lane = tid & 31;
    const int wave = tid >> 5;
    const int m0 = blockIdx.y * BM;
    const int n0 = blockIdx.x * BN;
    const int wm = wave & 3;   // rows 32*wm..+31 (2 tile rows)
    const int wn = wave >> 2;  // cols 64*wn..+63 (4 tile cols)
    const int lmod   = lane & 15;
    const int lhalf  = (lane < 16) ? 0 : 8;
    const int rowoff = (lane < 16) ? 0 : 8;

    auto stage = [&](int bufb, int k0) {
        #pragma unroll
        for (int p = 0; p < 4; ++p) {
            const int c = tid + 256 * p;          // 1024 16B chunks per tile
            const int row = c >> 3;
            const int cv  = (c & 7) * 8;
            cp16(A  + (size_t)(m0 + row) * K + k0 + cv, &As[bufb][row][cv]);
            cp16(BT + (size_t)(n0 + row) * K + k0 + cv, &Bs[bufb][row][cv]);
        }
    };

    v8f acc[2][4] = {};
    const int nk = K / BK;
    stage(0, 0);
    int buf = 0;

    for (int it = 0; it < nk; ++it) {
        stage_fence();                               // buf ready, prev reads done
        if (it + 1 < nk) stage(buf ^ 1, (it + 1) * BK);

        #pragma unroll
        for (int kk = 0; kk < 2; ++kk) {             // two 32-wide WMMA k-slices
            v16h af[2], bf[4];
            #pragma unroll
            for (int tm = 0; tm < 2; ++tm) {
                const int m = 32 * wm + 16 * tm + lmod;
                v8h lo = *reinterpret_cast<const v8h*>(&As[buf][m][32 * kk + lhalf]);
                v8h hi = *reinterpret_cast<const v8h*>(&As[buf][m][32 * kk + lhalf + 16]);
                af[tm] = cat16(lo, hi);
            }
            #pragma unroll
            for (int tn = 0; tn < 4; ++tn) {
                const int n = 64 * wn + 16 * tn + lmod;
                v8h lo = *reinterpret_cast<const v8h*>(&Bs[buf][n][32 * kk + lhalf]);
                v8h hi = *reinterpret_cast<const v8h*>(&Bs[buf][n][32 * kk + lhalf + 16]);
                bf[tn] = cat16(lo, hi);
            }
            #pragma unroll
            for (int tm = 0; tm < 2; ++tm)
                #pragma unroll
                for (int tn = 0; tn < 4; ++tn)
                    acc[tm][tn] = __builtin_amdgcn_wmma_f32_16x16x32_f16(
                        false, af[tm], false, bf[tn], (short)0, acc[tm][tn], false, false);
        }
        buf ^= 1;
    }

    #pragma unroll
    for (int tm = 0; tm < 2; ++tm) {
        #pragma unroll
        for (int tn = 0; tn < 4; ++tn) {
            const int col = n0 + 64 * wn + 16 * tn + lmod;
            const float bv = bias[col];
            #pragma unroll
            for (int r = 0; r < 8; ++r) {
                const int row = m0 + 32 * wm + 16 * tm + r + rowoff;
                float v = acc[tm][tn][r] + bv;
                if (RELU) v = fmaxf(v, 0.0f);
                C[(size_t)row * N + col] = (OT)v;
            }
        }
    }
}

// ---------------------------------------------------------------------------
// Flash attention: block = (b, h, 128 queries), 8 waves x 16 query rows.
// K tiles [32key][64d] and pre-transposed V tiles [64d][32key] staged async,
// double-buffered. floor(s/8) quirk before online softmax (DPP reductions).
// Fragment loads are batched ahead of the WMMA groups to amortize waits.
// ---------------------------------------------------------------------------
__global__ __launch_bounds__(256)
void wmma_attn_kernel(const _Float16* __restrict__ Q, const _Float16* __restrict__ Kg,
                      const _Float16* __restrict__ VT, float* __restrict__ O,
                      int S, int H, int HD)
{
    const int D = H * HD;
    constexpr int KT = 32;
    __shared__ alignas(16) _Float16 Ks[2][KT][72];     // [key][d]
    __shared__ alignas(16) _Float16 VTs[2][64][40];    // [d][key]
    __shared__ alignas(16) _Float16 Ps[8][16][40];     // per-wave P tile 16x32

    const int tid  = threadIdx.x;
    const int lane = tid & 31;
    const int wave = tid >> 5;
    const int lmod   = lane & 15;
    const int lhalf  = (lane < 16) ? 0 : 8;
    const int rowoff = (lane < 16) ? 0 : 8;

    const int q0 = blockIdx.x * 128;
    const int h  = blockIdx.y;
    const int b  = blockIdx.z;

    v16h qf[2];
    {
        const size_t qrow = (size_t)(b * S + q0 + 16 * wave + lmod) * D + h * HD;
        #pragma unroll
        for (int f = 0; f < 2; ++f) {
            v8h lo = *reinterpret_cast<const v8h*>(Q + qrow + 32 * f + lhalf);
            v8h hi = *reinterpret_cast<const v8h*>(Q + qrow + 32 * f + lhalf + 16);
            qf[f] = cat16(lo, hi);
        }
    }

    auto stage = [&](int bufb, int kv0) {
        {
            const int key = tid >> 3, cv = (tid & 7) * 8;          // 32x64 halves
            cp16(Kg + (size_t)(b * S + kv0 + key) * D + h * HD + cv, &Ks[bufb][key][cv]);
        }
        {
            const int d = tid >> 2, cv = (tid & 3) * 8;            // 64x32 halves
            cp16(VT + ((size_t)(b * H + h) * HD + d) * S + kv0 + cv, &VTs[bufb][d][cv]);
        }
    };

    v8f o_acc[4] = {};
    float m_i[8], l_i[8];
    #pragma unroll
    for (int r = 0; r < 8; ++r) { m_i[r] = -1e30f; l_i[r] = 0.0f; }

    stage(0, 0);
    int buf = 0;
    for (int kv0 = 0; kv0 < S; kv0 += KT) {
        stage_fence();
        if (kv0 + KT < S) stage(buf ^ 1, kv0 + KT);

        // ---- batch all K fragments, then 4 interleaved WMMAs ----
        v16h kf[2][2];                               // [key subtile][k-slice]
        #pragma unroll
        for (int t = 0; t < 2; ++t) {
            #pragma unroll
            for (int f = 0; f < 2; ++f) {
                v8h lo = *reinterpret_cast<const v8h*>(&Ks[buf][16 * t + lmod][32 * f + lhalf]);
                v8h hi = *reinterpret_cast<const v8h*>(&Ks[buf][16 * t + lmod][32 * f + lhalf + 16]);
                kf[t][f] = cat16(lo, hi);
            }
        }
        v8f s0 = {}, s1 = {};
        s0 = __builtin_amdgcn_wmma_f32_16x16x32_f16(false, qf[0], false, kf[0][0], (short)0, s0, false, false);
        s1 = __builtin_amdgcn_wmma_f32_16x16x32_f16(false, qf[0], false, kf[1][0], (short)0, s1, false, false);
        s0 = __builtin_amdgcn_wmma_f32_16x16x32_f16(false, qf[1], false, kf[0][1], (short)0, s0, false, false);
        s1 = __builtin_amdgcn_wmma_f32_16x16x32_f16(false, qf[1], false, kf[1][1], (short)0, s1, false, false);

        #pragma unroll
        for (int r = 0; r < 8; ++r) {
            s0[r] = floorf(s0[r] * 0.125f);    // faithful floor(s/sqrt(64)) quirk
            s1[r] = floorf(s1[r] * 0.125f);
        }

        float p0[8], p1[8], corr[8];
        #pragma unroll
        for (int r = 0; r < 8; ++r) {
            const float mx = row16_max(fmaxf(s0[r], s1[r]));
            const float m_new = fmaxf(m_i[r], mx);
            corr[r] = __expf(m_i[r] - m_new);
            p0[r] = __expf(s0[r] - m_new);
            p1[r] = __expf(s1[r] - m_new);
            const float rs = row16_sum(p0[r] + p1[r]);
            l_i[r] = l_i[r] * corr[r] + rs;
            m_i[r] = m_new;
        }
        #pragma unroll
        for (int dt = 0; dt < 4; ++dt)
            #pragma unroll
            for (int r = 0; r < 8; ++r)
                o_acc[dt][r] *= corr[r];

        // ---- batch V fragments first (independent of P), then P round-trip ----
        v16h vf[4];
        #pragma unroll
        for (int dt = 0; dt < 4; ++dt) {
            v8h lo = *reinterpret_cast<const v8h*>(&VTs[buf][16 * dt + lmod][lhalf]);
            v8h hi = *reinterpret_cast<const v8h*>(&VTs[buf][16 * dt + lmod][lhalf + 16]);
            vf[dt] = cat16(lo, hi);
        }
        #pragma unroll
        for (int r = 0; r < 8; ++r) {
            Ps[wave][r + rowoff][lmod]      = (_Float16)p0[r];
            Ps[wave][r + rowoff][16 + lmod] = (_Float16)p1[r];
        }
        v16h pf;    // DS ops are in-order within a wave: reads see the stores above
        {
            v8h lo = *reinterpret_cast<const v8h*>(&Ps[wave][lmod][lhalf]);
            v8h hi = *reinterpret_cast<const v8h*>(&Ps[wave][lmod][lhalf + 16]);
            pf = cat16(lo, hi);
        }

        // ---- O += P * V (4 independent d-tiles, contraction over 32 keys) ----
        #pragma unroll
        for (int dt = 0; dt < 4; ++dt)
            o_acc[dt] = __builtin_amdgcn_wmma_f32_16x16x32_f16(
                false, pf, false, vf[dt], (short)0, o_acc[dt], false, false);
        buf ^= 1;
    }

    #pragma unroll
    for (int r = 0; r < 8; ++r) {
        const float inv = 1.0f / l_i[r];
        const size_t row = (size_t)(b * S + q0 + 16 * wave + r + rowoff) * D + h * HD;
        #pragma unroll
        for (int dt = 0; dt < 4; ++dt)
            O[row + 16 * dt + lmod] = o_acc[dt][r] * inv;
    }
}

// ---------------------------------------------------------------------------
// Out = LN(A + Bres)*g + be ; optional f16 shadow copy for the next GEMM.
// ---------------------------------------------------------------------------
__global__ __launch_bounds__(256)
void add_ln_kernel(const float* __restrict__ A, const float* __restrict__ Bres,
                   const float* __restrict__ g, const float* __restrict__ be,
                   float* __restrict__ Out, _Float16* __restrict__ OutH)
{
    constexpr int D = 1024;
    const int row = blockIdx.x;
    const int tid = threadIdx.x;
    __shared__ float ws_[8], wq_[8];

    float vals[4], s = 0.0f, sq = 0.0f;
    #pragma unroll
    for (int i = 0; i < 4; ++i) {
        const int c = tid + i * 256;
        const float v = A[(size_t)row * D + c] + Bres[(size_t)row * D + c];
        vals[i] = v; s += v; sq += v * v;
    }
    #pragma unroll
    for (int m = 1; m < 32; m <<= 1) { s += __shfl_xor(s, m, 32); sq += __shfl_xor(sq, m, 32); }
    if ((tid & 31) == 0) { ws_[tid >> 5] = s; wq_[tid >> 5] = sq; }
    __syncthreads();
    s = 0.0f; sq = 0.0f;
    #pragma unroll
    for (int w = 0; w < 8; ++w) { s += ws_[w]; sq += wq_[w]; }
    const float mu  = s * (1.0f / D);
    const float var = sq * (1.0f / D) - mu * mu;
    const float rs  = rsqrtf(var + 1e-5f);
    #pragma unroll
    for (int i = 0; i < 4; ++i) {
        const int c = tid + i * 256;
        const float o = (vals[i] - mu) * rs * g[c] + be[c];
        Out[(size_t)row * D + c] = o;
        if (OutH) OutH[(size_t)row * D + c] = (_Float16)o;
    }
}

// ---------------------------------------------------------------------------
// One-time prep kernels.
// ---------------------------------------------------------------------------
__global__ __launch_bounds__(256)
void cvt_f16_kernel(const float* __restrict__ in, _Float16* __restrict__ out)
{
    const size_t i = ((size_t)blockIdx.x * 256 + threadIdx.x) * 4;
    const float4 v = *reinterpret_cast<const float4*>(in + i);
    out[i + 0] = (_Float16)v.x;
    out[i + 1] = (_Float16)v.y;
    out[i + 2] = (_Float16)v.z;
    out[i + 3] = (_Float16)v.w;
}

// out[n][k] = (f16) in[k][n] for in [K][N] f32
__global__ __launch_bounds__(256)
void transpose_cvt_kernel(const float* __restrict__ in, _Float16* __restrict__ out,
                          int K, int N)
{
    __shared__ float t[32][33];
    const int n0 = blockIdx.x * 32, k0 = blockIdx.y * 32;
    const int tx = threadIdx.x & 31, ty = threadIdx.x >> 5;
    #pragma unroll
    for (int i = 0; i < 32; i += 8)
        t[ty + i][tx] = in[(size_t)(k0 + ty + i) * N + n0 + tx];
    __syncthreads();
    #pragma unroll
    for (int i = 0; i < 32; i += 8)
        out[(size_t)(n0 + ty + i) * K + k0 + tx] = (_Float16)t[tx][ty + i];
}

// vT[b,h,d,s] = v[b,s,h*HD+d]   (f16 -> f16)
__global__ __launch_bounds__(256)
void transpose_v_kernel(const _Float16* __restrict__ in, _Float16* __restrict__ out)
{
    constexpr int S = 2048, H = 16, HD = 64, D = 1024;
    __shared__ _Float16 t[32][33];
    const int bh = blockIdx.z, b = bh >> 4, h = bh & 15;
    const int s0 = blockIdx.x * 32, d0 = blockIdx.y * 32;
    const int tx = threadIdx.x & 31, ty = threadIdx.x >> 5;
    #pragma unroll
    for (int i = 0; i < 32; i += 8)
        t[ty + i][tx] = in[(size_t)(b * S + s0 + ty + i) * D + h * HD + d0 + tx];
    __syncthreads();
    #pragma unroll
    for (int i = 0; i < 32; i += 8)
        out[((size_t)(b * H + h) * HD + d0 + ty + i) * S + s0 + tx] = t[tx][ty + i];
}

// ---------------------------------------------------------------------------
extern "C" void kernel_launch(void* const* d_in, const int* in_sizes, int n_in,
                              void* d_out, int out_size, void* d_ws, size_t ws_size,
                              hipStream_t stream)
{
    constexpr int Bn = 2, S = 2048, D = 1024, H = 16, HD = 64;
    const int M = Bn * S;                        // 4096
    const size_t BSD = (size_t)Bn * S * D;       // 4 Mi elements

    const float* x   = (const float*)d_in[0];
    const float* wq  = (const float*)d_in[1];
    const float* bq  = (const float*)d_in[2];
    const float* wk  = (const float*)d_in[3];
    const float* bk  = (const float*)d_in[4];
    const float* wv  = (const float*)d_in[5];
    const float* bv  = (const float*)d_in[6];
    const float* w1  = (const float*)d_in[7];
    const float* b1  = (const float*)d_in[8];
    const float* w2  = (const float*)d_in[9];
    const float* b2  = (const float*)d_in[10];
    const float* g1  = (const float*)d_in[11];
    const float* be1 = (const float*)d_in[12];
    const float* g2  = (const float*)d_in[13];
    const float* be2 = (const float*)d_in[14];
    float* out = (float*)d_out;

    // workspace layout (with aliasing to cap peak at ~102 MB)
    char* w = (char*)d_ws;
    _Float16* xh  = (_Float16*)w; w += BSD * 2;                     // 8 MB
    _Float16* wqT = (_Float16*)w; w += (size_t)D * D * 2;           // 2 MB
    _Float16* wkT = (_Float16*)w; w += (size_t)D * D * 2;
    _Float16* wvT = (_Float16*)w; w += (size_t)D * D * 2;
    _Float16* w1T = (_Float16*)w; w += (size_t)D * 4 * D * 2;       // 8 MB
    _Float16* w2T = (_Float16*)w; w += (size_t)D * 4 * D * 2;       // 8 MB
    char* qkv_region = w;                                           // 32 MB
    _Float16* q_h = (_Float16*)w; w += BSD * 2;
    _Float16* k_h = (_Float16*)w; w += BSD * 2;
    _Float16* v_h = (_Float16*)w; w += BSD * 2;
    _Float16* vT  = (_Float16*)w; w += BSD * 2;
    float* sa = (float*)w; w += BSD * 4;                            // 16 MB
    float* x1 = (float*)w; w += BSD * 4;                            // 16 MB
    _Float16* x1h = (_Float16*)w; w += BSD * 2;                     // 8 MB
    _Float16* h1 = (_Float16*)qkv_region;   // aliases q/k/v/vT (dead after attn)
    float*    ff = sa;                      // aliases sa (dead after LN1)

    const dim3 blk(256);

    // ---- one-time prep: f16 activations + transposed f16 weights ----
    cvt_f16_kernel<<<dim3((unsigned)(BSD / 1024)), blk, 0, stream>>>(x, xh);
    transpose_cvt_kernel<<<dim3(D / 32, D / 32), blk, 0, stream>>>(wq, wqT, D, D);
    transpose_cvt_kernel<<<dim3(D / 32, D / 32), blk, 0, stream>>>(wk, wkT, D, D);
    transpose_cvt_kernel<<<dim3(D / 32, D / 32), blk, 0, stream>>>(wv, wvT, D, D);
    transpose_cvt_kernel<<<dim3(4 * D / 32, D / 32), blk, 0, stream>>>(w1, w1T, D, 4 * D);
    transpose_cvt_kernel<<<dim3(D / 32, 4 * D / 32), blk, 0, stream>>>(w2, w2T, 4 * D, D);

    // ---- QKV projections ----
    wmma_gemm_f16<false, _Float16>
        <<<dim3(D / 128, M / 128), blk, 0, stream>>>(xh, wqT, bq, q_h, M, D, D);
    wmma_gemm_f16<false, _Float16>
        <<<dim3(D / 128, M / 128), blk, 0, stream>>>(xh, wkT, bk, k_h, M, D, D);
    wmma_gemm_f16<false, _Float16>
        <<<dim3(D / 128, M / 128), blk, 0, stream>>>(xh, wvT, bv, v_h, M, D, D);
    transpose_v_kernel<<<dim3(S / 32, HD / 32, Bn * H), blk, 0, stream>>>(v_h, vT);

    // ---- attention ----
    wmma_attn_kernel<<<dim3(S / 128, H, Bn), blk, 0, stream>>>(q_h, k_h, vT, sa, S, H, HD);

    // ---- x1 = LN(x + sa) (+ f16 shadow) ----
    add_ln_kernel<<<dim3(M), blk, 0, stream>>>(x, sa, g1, be1, x1, x1h);

    // ---- FFN (double-ReLU quirk) ----
    wmma_gemm_f16<true, _Float16>
        <<<dim3(4 * D / 128, M / 128), blk, 0, stream>>>(x1h, w1T, b1, h1, M, 4 * D, D);
    wmma_gemm_f16<true, float>
        <<<dim3(D / 128, M / 128), blk, 0, stream>>>(h1, w2T, b2, ff, M, D, 4 * D);

    // ---- out = LN(x1 + ff) ----
    add_ln_kernel<<<dim3(M), blk, 0, stream>>>(x1, ff, g2, be2, out, nullptr);
}
